// block_large_quantized_55877524521308
// MI455X (gfx1250) — matrix operand order repaired and checked by reading, not compiled, not run on hardware
//
#include <hip/hip_runtime.h>
#include <hip/hip_bf16.h>

// ---------------------------------------------------------------------------
// CDNA5 (gfx1250) bottleneck block via WMMA f16 GEMMs, wave32.
//   conv1: 1x1 256->64   GEMM M=64,K=256,N=100352
//   conv2: 3x3 64->64    GEMM M=64,K=576,N=100352 (9 taps, zero pad fused)
//   conv3: 1x1 64->256   GEMM M=256,K=64,N=100352
// BN(training) stats computed per-channel between convs; normalize+ReLU fused
// into the consumer's B-tile load. Weights fake-quantized and pre-swizzled
// into WMMA A-fragment layout; each block stages its A-fragment region into
// LDS once with GLOBAL_LOAD_ASYNC_TO_LDS_B128 (ASYNCcnt). ~129 MB workspace.
// ---------------------------------------------------------------------------

typedef __attribute__((ext_vector_type(16))) _Float16 v16h;
typedef __attribute__((ext_vector_type(8)))  _Float16 v8h;
typedef __attribute__((ext_vector_type(8)))  float    v8f;

#define PLANE 100352   // 32 * 3136 pixels per channel
#define IMG   3136     // 56*56
#define WDIM  56

__device__ __forceinline__ void async_wait0() {
#if __has_builtin(__builtin_amdgcn_s_wait_asynccnt)
  __builtin_amdgcn_s_wait_asynccnt(0);
#else
  asm volatile("s_wait_asynccnt 0x0" ::: "memory");
#endif
}

// ---------------------------------------------------------------------------
// Per-out-channel fake-quant scales: scale = max(amax/127, 1e-8)
// blocks 0..63 -> w1 (len 256), 64..127 -> w2 (len 576), 128..383 -> w3 (len 64)
// ---------------------------------------------------------------------------
__global__ __launch_bounds__(128) void wscale_kernel(
    const float* __restrict__ w1, const float* __restrict__ w2,
    const float* __restrict__ w3, float* __restrict__ scales) {
  int b = blockIdx.x;
  const float* src; int len;
  if (b < 64)       { src = w1 + b * 256;        len = 256; }
  else if (b < 128) { src = w2 + (b - 64) * 576; len = 576; }
  else              { src = w3 + (b - 128) * 64; len = 64; }
  float m = 0.f;
  for (int i = threadIdx.x; i < len; i += 128) m = fmaxf(m, fabsf(src[i]));
  __shared__ float red[128];
  red[threadIdx.x] = m;
  __syncthreads();
  for (int off = 64; off > 0; off >>= 1) {
    if (threadIdx.x < off)
      red[threadIdx.x] = fmaxf(red[threadIdx.x], red[threadIdx.x + off]);
    __syncthreads();
  }
  if (threadIdx.x == 0) scales[b] = fmaxf(red[0] * (1.0f / 127.0f), 1e-8f);
}

// ---------------------------------------------------------------------------
// Quantize weights and scatter into WMMA A-fragment layout:
//   frag[(mo*KT + kt)*512 + lane*16 + e] = Wq[M = mo*16 + (lane&15)]
//                                            [K = kt*32 + klocal(lane,e)]
//   klocal = (e&7) | ((e&8)<<1) | ((lane&16)>>1)    (16-bit A layout, ISA 7.12.2)
// K enumeration: conv1: K=c (KT=8); conv2: K=tap*64+c (KT=18); conv3: K=c (KT=2)
// ---------------------------------------------------------------------------
__global__ __launch_bounds__(256) void wfrag_kernel(
    const float* __restrict__ w1, const float* __restrict__ w2,
    const float* __restrict__ w3, const float* __restrict__ scales,
    _Float16* __restrict__ f1, _Float16* __restrict__ f2,
    _Float16* __restrict__ f3) {
  int t = blockIdx.x * 256 + threadIdx.x;
  if (t >= 69632) return;
  _Float16* dst; int KT, scOff, which, local;
  if (t < 16384)      { local = t;          dst = f1; KT = 8;  scOff = 0;   which = 1; }
  else if (t < 53248) { local = t - 16384;  dst = f2; KT = 18; scOff = 64;  which = 2; }
  else                { local = t - 53248;  dst = f3; KT = 2;  scOff = 128; which = 3; }
  int e = local & 15;
  int l = (local >> 4) & 31;
  int tile = local >> 9;
  int kt = tile % KT;
  int mo = tile / KT;
  int M = mo * 16 + (l & 15);
  int klocal = (e & 7) | ((e & 8) << 1) | ((l & 16) >> 1);
  int K = kt * 32 + klocal;
  float wv;
  if (which == 1) {
    wv = w1[M * 256 + K];
  } else if (which == 2) {
    int tap = K / 64, c = K % 64;            // OIHW: w2[M][c][tap]
    wv = w2[(M * 64 + c) * 9 + tap];
  } else {
    wv = w3[M * 64 + K];
  }
  float sc = scales[scOff + M];
  float q = fminf(fmaxf(rintf(wv / sc), -127.f), 127.f) * sc;
  dst[local] = (_Float16)q;
}

// ---------------------------------------------------------------------------
// GEMM conv kernel. Block = 128 threads = 4 waves; tile = 64 out-ch x 64 pix.
// Wave w computes rows [w*16, w*16+16) across four 16-pixel sub-tiles.
// A fragments for the whole block staged once into LDS via async copy;
// B tiles staged to LDS as [pix][k] (stride 40 f16 => 16B-aligned frag loads).
// ---------------------------------------------------------------------------
template <int CIN, int TAPS, bool IN_IS_F32, bool FUSE_BN, bool OUT_F32>
__global__ __launch_bounds__(128) void conv_gemm_kernel(
    const void* __restrict__ inAny,          // f32 NCHW (conv1) or f16 [c][p]
    const _Float16* __restrict__ wfrag,      // pre-swizzled A fragments
    const float* __restrict__ bnp,           // (scale,shift) pairs per in-ch
    void* __restrict__ outAny) {             // f16 [c][p] or f32 [c][p]
  constexpr int KT = TAPS * (CIN / 32);
  const int tid = threadIdx.x;
  const int wave = tid >> 5;
  const int lane = tid & 31;
  const int pixbase = blockIdx.x * 64;       // 3136 % 64 == 0 -> one image
  const int n = pixbase / IMG;
  const int s0 = pixbase - n * IMG;

  __shared__ __align__(32) _Float16 lds_b[64 * 40];
  __shared__ __align__(16) _Float16 lds_a[KT * 2048];  // 4 waves * KT * 512 f16

  // ---- stage this block's A-fragment region into LDS (pure byte copy) ----
  // GLOBAL_LOAD_ASYNC_TO_LDS_B128: per-lane 16B global->LDS, ASYNCcnt tracked.
  {
    const char* gsrc = (const char*)(wfrag + (size_t)blockIdx.y * (4 * KT * 512));
    const unsigned lbase = (unsigned)(uintptr_t)(void*)&lds_a[0];  // LDS byte offset
    constexpr int CHUNKS = (KT * 4096) / 16;   // multiples of 128 for all convs
#pragma unroll 4
    for (int i = tid; i < CHUNKS; i += 128) {
      unsigned loff = lbase + (unsigned)i * 16u;
      unsigned long long gaddr = (unsigned long long)(uintptr_t)(gsrc + (size_t)i * 16);
      asm volatile("global_load_async_to_lds_b128 %0, %1, off"
                   :: "v"(loff), "v"(gaddr)
                   : "memory");
    }
    async_wait0();
  }
  __syncthreads();

  v8f acc[4] = {};

  const int kStage = tid >> 2;               // 0..31  (k row this thread stages)
  const int pxo    = (tid & 3) * 16;         // 0,16,32,48

  for (int tap = 0; tap < TAPS; ++tap) {
    const int dy = (TAPS == 9) ? (tap / 3 - 1) : 0;
    const int dx = (TAPS == 9) ? (tap % 3 - 1) : 0;
    for (int ct = 0; ct < CIN / 32; ++ct) {
      __syncthreads();                       // prior compute done
      // ---- stage B tile: 32 k x 64 pixels, f16, LDS[pix][k] ----
      {
        const int c = ct * 32 + kStage;
        if constexpr (IN_IS_F32) {
          // conv1: x is NCHW fp32, plane address ((n*256 + c)*3136 + s)
          const float* src =
              (const float*)inAny + (size_t)(n * 256 + c) * IMG + s0 + pxo;
#pragma unroll
          for (int j4 = 0; j4 < 4; ++j4) {
            float4 v = ((const float4*)src)[j4];
            int pj = pxo + j4 * 4;
            lds_b[(pj + 0) * 40 + kStage] = (_Float16)v.x;
            lds_b[(pj + 1) * 40 + kStage] = (_Float16)v.y;
            lds_b[(pj + 2) * 40 + kStage] = (_Float16)v.z;
            lds_b[(pj + 3) * 40 + kStage] = (_Float16)v.w;
          }
        } else {
          const _Float16* hp =
              (const _Float16*)inAny + (size_t)c * PLANE + (size_t)n * IMG;
          float sc = 1.f, sh = 0.f;
          if constexpr (FUSE_BN) { sc = bnp[2 * c]; sh = bnp[2 * c + 1]; }
#pragma unroll
          for (int j = 0; j < 16; ++j) {
            int s = s0 + pxo + j;
            float v;
            if constexpr (TAPS == 9) {
              int y = s / WDIM + dy;
              int x = s % WDIM + dx;
              bool ok = ((unsigned)y < (unsigned)WDIM) &&
                        ((unsigned)x < (unsigned)WDIM);
              // zero padding is applied AFTER relu(bn(.)) in the reference
              v = ok ? fmaxf((float)hp[y * WDIM + x] * sc + sh, 0.f) : 0.f;
            } else {
              float raw = (float)hp[s];
              v = FUSE_BN ? fmaxf(raw * sc + sh, 0.f) : raw;
            }
            lds_b[(pxo + j) * 40 + kStage] = (_Float16)v;
          }
        }
      }
      __syncthreads();                       // tile staged

      // ---- WMMA compute: A from LDS, B from LDS ----
      const int ktIdx = tap * (CIN / 32) + ct;
      const v8h* ap =
          (const v8h*)&lds_a[((wave * KT + ktIdx) << 9) + lane * 16];
      v8h alo = ap[0];
      v8h ahi = ap[1];
      v16h af;
#pragma unroll
      for (int i = 0; i < 8; ++i) { af[i] = alo[i]; af[i + 8] = ahi[i]; }

      const int prow = lane & 15;
      const int kb = (lane < 16) ? 0 : 8;
#pragma unroll
      for (int sub = 0; sub < 4; ++sub) {
        const v8h* bp = (const v8h*)&lds_b[(sub * 16 + prow) * 40 + kb];
        v8h blo = bp[0];
        v8h bhi = bp[2];
        v16h bf;
#pragma unroll
        for (int i = 0; i < 8; ++i) { bf[i] = blo[i]; bf[i + 8] = bhi[i]; }
        acc[sub] = __builtin_amdgcn_wmma_f32_16x16x32_f16(
            false, af, false, bf, (short)0, acc[sub], false, false);
      }
    }
  }

  // ---- epilogue: D layout lane -> (col = lane&15, rows r + 8*(lane>=16)) ----
  const int chBase = blockIdx.y * 64 + wave * 16 + ((lane >> 4) << 3);
  const int pix = pixbase + (lane & 15);
#pragma unroll
  for (int sub = 0; sub < 4; ++sub) {
    int p = pix + sub * 16;
#pragma unroll
    for (int r = 0; r < 8; ++r) {
      float v = acc[sub][r];
      size_t o = (size_t)(chBase + r) * PLANE + p;
      if constexpr (OUT_F32) ((float*)outAny)[o] = v;
      else                   ((_Float16*)outAny)[o] = (_Float16)v;
    }
  }
}

// ---------------------------------------------------------------------------
// Per-channel BN statistics over the 100352-pixel plane -> (scale, shift)
// scale = gamma * rsqrt(var+eps); shift = beta - mean*scale
// ---------------------------------------------------------------------------
template <bool IN_F32>
__global__ __launch_bounds__(256) void bn_stats_kernel(
    const void* __restrict__ hAny, const float* __restrict__ gamma,
    const float* __restrict__ beta, float* __restrict__ params) {
  const int c = blockIdx.x;
  const size_t base = (size_t)c * PLANE;
  float s = 0.f, sq = 0.f;
  for (int i = threadIdx.x; i < PLANE; i += 256) {
    float v = IN_F32 ? ((const float*)hAny)[base + i]
                     : (float)((const _Float16*)hAny)[base + i];
    s += v;
    sq += v * v;
  }
  __shared__ float rs[256];
  __shared__ float rq[256];
  rs[threadIdx.x] = s;
  rq[threadIdx.x] = sq;
  __syncthreads();
  for (int off = 128; off > 0; off >>= 1) {
    if (threadIdx.x < off) {
      rs[threadIdx.x] += rs[threadIdx.x + off];
      rq[threadIdx.x] += rq[threadIdx.x + off];
    }
    __syncthreads();
  }
  if (threadIdx.x == 0) {
    const float inv_n = 1.0f / (float)PLANE;
    float mean = rs[0] * inv_n;
    float var = rq[0] * inv_n - mean * mean;
    float inv = rsqrtf(var + 1e-5f);
    float sc = gamma[c] * inv;
    params[2 * c] = sc;
    params[2 * c + 1] = beta[c] - mean * sc;
  }
}

// ---------------------------------------------------------------------------
// out[n][c][s] = relu( h3[c][n*3136+s]*sc3[c] + sh3[c] + x[n][c][s] )
// Fully coalesced on both sides ([c][n][s] <-> [n][c][s] plane transpose).
// ---------------------------------------------------------------------------
__global__ __launch_bounds__(256) void fuse_out_kernel(
    const float* __restrict__ h3, const float* __restrict__ x,
    const float* __restrict__ p3, float* __restrict__ out) {
  unsigned id = blockIdx.x * 256u + threadIdx.x;   // < 25,690,112
  unsigned c = id / PLANE;
  unsigned pp = id - c * PLANE;
  unsigned n = pp / IMG;
  unsigned s = pp - n * IMG;
  float v = h3[id] * p3[2 * c] + p3[2 * c + 1];
  size_t o = (size_t)(n * 256u + c) * IMG + s;
  v += x[o];
  out[o] = fmaxf(v, 0.f);
}

// ---------------------------------------------------------------------------
extern "C" void kernel_launch(void* const* d_in, const int* in_sizes, int n_in,
                              void* d_out, int out_size, void* d_ws,
                              size_t ws_size, hipStream_t stream) {
  const float* x  = (const float*)d_in[0];
  const float* w1 = (const float*)d_in[1];
  const float* g1 = (const float*)d_in[2];
  const float* b1 = (const float*)d_in[3];
  const float* w2 = (const float*)d_in[4];
  const float* g2 = (const float*)d_in[5];
  const float* b2 = (const float*)d_in[6];
  const float* w3 = (const float*)d_in[7];
  const float* g3 = (const float*)d_in[8];
  const float* b3 = (const float*)d_in[9];

  char* ws = (char*)d_ws;
  _Float16* f1    = (_Float16*)(ws + 0);          //  32768 B
  _Float16* f2    = (_Float16*)(ws + 32768);      //  73728 B
  _Float16* f3    = (_Float16*)(ws + 106496);     //  32768 B
  float* scales   = (float*)(ws + 139264);        //   1536 B
  float* p1       = (float*)(ws + 140800);        //    512 B
  float* p2       = (float*)(ws + 141312);        //    512 B
  float* p3       = (float*)(ws + 141824);        //   2048 B
  _Float16* h1    = (_Float16*)(ws + 147456);     // 12,845,056 B
  _Float16* h2    = (_Float16*)(ws + 12992512);   // 12,845,056 B
  float*    h3    = (float*)(ws + 25837568);      // 102,760,448 B (ends ~128.6MB)

  // 1. fake-quant scales + weight fragments (WMMA A layout)
  wscale_kernel<<<384, 128, 0, stream>>>(w1, w2, w3, scales);
  wfrag_kernel<<<272, 256, 0, stream>>>(w1, w2, w3, scales, f1, f2, f3);

  // 2. conv1 (1x1, 256->64), raw output h1 (f16 [c][p])
  conv_gemm_kernel<256, 1, true, false, false>
      <<<dim3(1568, 1), 128, 0, stream>>>(x, f1, nullptr, h1);
  bn_stats_kernel<false><<<64, 256, 0, stream>>>(h1, g1, b1, p1);

  // 3. conv2 (3x3 pad1, 64->64), BN1+relu fused into B loads
  conv_gemm_kernel<64, 9, false, true, false>
      <<<dim3(1568, 1), 128, 0, stream>>>(h1, f2, p1, h2);
  bn_stats_kernel<false><<<64, 256, 0, stream>>>(h2, g2, b2, p2);

  // 4. conv3 (1x1, 64->256), BN2+relu fused, f32 output h3
  conv_gemm_kernel<64, 1, false, true, true>
      <<<dim3(1568, 4), 128, 0, stream>>>(h2, f3, p2, h3);
  bn_stats_kernel<true><<<256, 256, 0, stream>>>(h3, g3, b3, p3);

  // 5. BN3 + residual + relu, coalesced plane transpose to NCHW
  fuse_out_kernel<<<100352, 256, 0, stream>>>(h3, x, p3, (float*)d_out);
}
